// xLSTM_88029649698943
// MI455X (gfx1250) — compile-verified, matrix operand order checked by reading, not run on hardware
//
#include <hip/hip_runtime.h>

// CDNA5 (gfx1250) fused 2-layer LSTM + MLP head.
// BN=9600 sequences, T=64 steps, H=64, gates=256. wave32, WMMA f16->f32.
// Recurrent weights live in LDS in WMMA-B fragment order; h/c state in LDS;
// gate nonlinearities are branch-free v_exp/v_rcp sequences.

typedef __attribute__((ext_vector_type(16))) _Float16 v16h;
typedef __attribute__((ext_vector_type(8)))  float    v8f;

#define BN      9600
#define SEQ_T   64
#define HID     64
#define G4      256          // 4*H
#define MTILE   16           // sequences per workgroup
#define NBLK    (BN / MTILE) // 600 workgroups

// ---- dynamic LDS layout (bytes) ----
#define OFF_WB0   0                          // [2*16][32*16] f16  (32768)
#define OFF_WB1   (OFF_WB0 + 32768)          // [4*16][32*16] f16  (65536)
#define OFF_A0    (OFF_WB1 + 65536)          // [16][64]  f16      (2048)
#define OFF_A1    (OFF_A0  + 2048)           // [16][128] f16      (4096)
#define OFF_G     (OFF_A1  + 4096)           // [16][256] f32      (16384)
#define OFF_C0    (OFF_G   + 16384)          // [16][64]  f32      (4096)
#define OFF_C1    (OFF_C0  + 4096)           // [16][64]  f32      (4096)
#define OFF_H1    (OFF_C1  + 4096)           // [16][64]  f32      (4096)
#define OFF_X     (OFF_H1  + 4096)           // [16][64]  f32      (4096)
#define OFF_B0    (OFF_X   + 4096)           // [256] f32          (1024)
#define OFF_B1    (OFF_B0  + 1024)           // [256] f32          (1024)
#define OFF_WI0   (OFF_B1  + 1024)           // [256] f32          (1024)
#define OFF_FC    (OFF_WI0 + 1024)           // [16][16] f32       (1024)
#define SMEM_BYTES (OFF_FC + 1024)           // = 141312

// Branch-free fast transcendentals: v_exp_f32 + v_rcp_f32, no libm divide,
// no EXEC divergence in the recurrent loop.
__device__ __forceinline__ float rcp_(float x) {
    return __builtin_amdgcn_rcpf(x);
}
__device__ __forceinline__ float sig_(float x) {
    return rcp_(1.0f + __expf(-x));
}
__device__ __forceinline__ float tanh_(float x) {
    x = fminf(fmaxf(x, -20.0f), 20.0f);     // keep exp(2x) finite
    float e = __expf(2.0f * x);
    return (e - 1.0f) * rcp_(e + 1.0f);
}

__global__ void lstm_fused_kernel(const float* __restrict__ x,
                                  const float* __restrict__ Wih0,
                                  const float* __restrict__ Whh0,
                                  const float* __restrict__ bih0,
                                  const float* __restrict__ bhh0,
                                  const float* __restrict__ Wih1,
                                  const float* __restrict__ Whh1,
                                  const float* __restrict__ bih1,
                                  const float* __restrict__ bhh1,
                                  const float* __restrict__ fc1_w,
                                  const float* __restrict__ fc1_b,
                                  const float* __restrict__ fc2_w,
                                  const float* __restrict__ fc2_b,
                                  float* __restrict__ out) {
    extern __shared__ unsigned char smem[];
    _Float16* sWB0 = (_Float16*)(smem + OFF_WB0);   // pre-swizzled B frags, layer0
    _Float16* sWB1 = (_Float16*)(smem + OFF_WB1);   // pre-swizzled B frags, layer1
    _Float16* sA0  = (_Float16*)(smem + OFF_A0);    // h0 (f16), row-major 16x64
    _Float16* sA1  = (_Float16*)(smem + OFF_A1);    // [h1 | h0out] (f16), 16x128
    float*    sG   = (float*)(smem + OFF_G);        // gate pre-activations 16x256
    float*    sC0  = (float*)(smem + OFF_C0);
    float*    sC1  = (float*)(smem + OFF_C1);
    float*    sH1  = (float*)(smem + OFF_H1);
    float*    sX   = (float*)(smem + OFF_X);
    float*    sB0  = (float*)(smem + OFF_B0);
    float*    sB1  = (float*)(smem + OFF_B1);
    float*    sWI0 = (float*)(smem + OFF_WI0);
    float*    sFC  = (float*)(smem + OFF_FC);

    const int tid  = threadIdx.x;     // 0..255
    const int seq0 = blockIdx.x * MTILE;

    // ---------------- weight staging (f32 global -> f16 LDS, WMMA-B swizzle) ----
    // B-fragment layout chosen here (contiguous k, same order as the A-frag reads):
    //   lane = ((k_in_tile>>4)&1)*16 + (n&15), elem i = k&15 -> contiguous 16 halves/lane.
    // Layer 0: B = Whh0^T (K=64 x N=256), src Whh0[g][k] row-major [256][64]
    for (int idx = tid; idx < G4 * HID; idx += 256) {
        int n = idx >> 6;            // gate col 0..255
        int k = idx & 63;            // K row 0..63
        float v = Whh0[n * HID + k];
        int kt = k >> 5, nt = n >> 4;
        int lane = ((k >> 4) & 1) * 16 + (n & 15);
        sWB0[((kt * 16 + nt) * 32 + lane) * 16 + (k & 15)] = (_Float16)v;
    }
    // Layer 1: B = [Whh1^T ; Wih1^T] (K=128 x N=256)
    for (int idx = tid; idx < G4 * 128; idx += 256) {
        int n = idx >> 7;            // gate col
        int k = idx & 127;           // 0..63 -> h1 part, 64..127 -> h0out part
        float v = (k < 64) ? Whh1[n * HID + k] : Wih1[n * HID + (k - 64)];
        int kt = k >> 5, nt = n >> 4;
        int lane = ((k >> 4) & 1) * 16 + (n & 15);
        sWB1[((kt * 16 + nt) * 32 + lane) * 16 + (k & 15)] = (_Float16)v;
    }
    // biases, scalar-input weights
    if (tid < G4) {
        sB0[tid]  = bih0[tid] + bhh0[tid];
        sB1[tid]  = bih1[tid] + bhh1[tid];
        sWI0[tid] = Wih0[tid];           // [256 x 1]
    }
    // stage the 16 sequences' inputs (each sequence is 64 contiguous floats)
    for (int idx = tid; idx < MTILE * SEQ_T; idx += 256) {
        int r = idx >> 6, t = idx & 63;
        sX[r * SEQ_T + t] = x[(size_t)(seq0 + r) * SEQ_T + t];
    }
    // zero state
    for (int idx = tid; idx < MTILE * HID; idx += 256) {
        int r = idx >> 6, h = idx & 63;
        sC0[r * HID + h] = 0.0f;
        sC1[r * HID + h] = 0.0f;
        sH1[r * HID + h] = 0.0f;
        sA0[r * HID + h]        = (_Float16)0.0f;
        sA1[r * 128 + h]        = (_Float16)0.0f;
        sA1[r * 128 + 64 + h]   = (_Float16)0.0f;
    }
    __syncthreads();

    const int wave = tid >> 5;
    const int lane = tid & 31;
    const int arow = lane & 15;           // A-frag row for this lane
    const int akb  = (lane >> 4) * 16;    // A-frag K base
    const int nt0  = wave * 2;
    const int nt1  = wave * 2 + 1;
    const int crow = (lane >> 4) * 8;     // C-frag row base
    const int ccol0 = nt0 * 16 + (lane & 15);
    const int ccol1 = nt1 * 16 + (lane & 15);

    // ---------------- recurrence over T=64 steps ----------------
    for (int t = 0; t < SEQ_T; ++t) {
        // --- layer 0 gates: h0[16x64] @ Whh0^T -> sG[16x256] ---
        {
            v8f acc0 = {}, acc1 = {};
            #pragma unroll
            for (int kt = 0; kt < 2; ++kt) {
                v16h a  = *(const v16h*)&sA0[arow * HID + kt * 32 + akb];
                v16h b0 = *(const v16h*)&sWB0[((kt * 16 + nt0) * 32 + lane) * 16];
                v16h b1 = *(const v16h*)&sWB0[((kt * 16 + nt1) * 32 + lane) * 16];
                acc0 = __builtin_amdgcn_wmma_f32_16x16x32_f16(false, a, false, b0,
                                                              (short)0, acc0, false, false);
                acc1 = __builtin_amdgcn_wmma_f32_16x16x32_f16(false, a, false, b1,
                                                              (short)0, acc1, false, false);
            }
            #pragma unroll
            for (int r = 0; r < 8; ++r) {
                sG[(crow + r) * G4 + ccol0] = acc0[r];
                sG[(crow + r) * G4 + ccol1] = acc1[r];
            }
        }
        __syncthreads();

        // --- layer 0 elementwise: add bias + x_t*Wih0, activate, update c0/h0 ---
        #pragma unroll
        for (int q = 0; q < 4; ++q) {
            int idx = tid + q * 256;          // 0..1023
            int r = idx >> 6, h = idx & 63;
            float xt = sX[r * SEQ_T + t];
            float gi = sG[r * G4 + h]         + sB0[h]         + xt * sWI0[h];
            float gf = sG[r * G4 + 64 + h]    + sB0[64 + h]    + xt * sWI0[64 + h];
            float gg = sG[r * G4 + 128 + h]   + sB0[128 + h]   + xt * sWI0[128 + h];
            float go = sG[r * G4 + 192 + h]   + sB0[192 + h]   + xt * sWI0[192 + h];
            float c  = sig_(gf) * sC0[r * HID + h] + sig_(gi) * tanh_(gg);
            float hn = sig_(go) * tanh_(c);
            sC0[r * HID + h] = c;
            sA0[r * HID + h]      = (_Float16)hn;   // next-step layer0 A
            sA1[r * 128 + 64 + h] = (_Float16)hn;   // layer1 input half of A
        }
        __syncthreads();

        // --- layer 1 gates: [h1|h0out][16x128] @ [Whh1;Wih1]^T -> sG ---
        {
            v8f acc0 = {}, acc1 = {};
            #pragma unroll
            for (int kt = 0; kt < 4; ++kt) {
                v16h a  = *(const v16h*)&sA1[arow * 128 + kt * 32 + akb];
                v16h b0 = *(const v16h*)&sWB1[((kt * 16 + nt0) * 32 + lane) * 16];
                v16h b1 = *(const v16h*)&sWB1[((kt * 16 + nt1) * 32 + lane) * 16];
                acc0 = __builtin_amdgcn_wmma_f32_16x16x32_f16(false, a, false, b0,
                                                              (short)0, acc0, false, false);
                acc1 = __builtin_amdgcn_wmma_f32_16x16x32_f16(false, a, false, b1,
                                                              (short)0, acc1, false, false);
            }
            #pragma unroll
            for (int r = 0; r < 8; ++r) {
                sG[(crow + r) * G4 + ccol0] = acc0[r];
                sG[(crow + r) * G4 + ccol1] = acc1[r];
            }
        }
        __syncthreads();

        // --- layer 1 elementwise update ---
        #pragma unroll
        for (int q = 0; q < 4; ++q) {
            int idx = tid + q * 256;
            int r = idx >> 6, h = idx & 63;
            float gi = sG[r * G4 + h]       + sB1[h];
            float gf = sG[r * G4 + 64 + h]  + sB1[64 + h];
            float gg = sG[r * G4 + 128 + h] + sB1[128 + h];
            float go = sG[r * G4 + 192 + h] + sB1[192 + h];
            float c  = sig_(gf) * sC1[r * HID + h] + sig_(gi) * tanh_(gg);
            float hn = sig_(go) * tanh_(c);
            sC1[r * HID + h] = c;
            sH1[r * HID + h] = hn;                  // f32 for output/MLP
            sA1[r * 128 + h] = (_Float16)hn;        // next-step layer1 recurrent A
        }
        __syncthreads();
    }

    // ---------------- epilogue: hT output + tiny MLP ----------------
    #pragma unroll
    for (int q = 0; q < 4; ++q) {
        int idx = tid + q * 256;
        int r = idx >> 6, h = idx & 63;
        out[(size_t)(seq0 + r) * HID + h] = sH1[r * HID + h];
    }
    // fc1: 16 rows x 16 outputs, input = [h1 (64) | x_last (1)]
    {
        int r = tid >> 4, j = tid & 15;
        float acc = fc1_b[j];
        #pragma unroll 8
        for (int k = 0; k < HID; ++k)
            acc += sH1[r * HID + k] * fc1_w[j * 65 + k];
        acc += sX[r * SEQ_T + (SEQ_T - 1)] * fc1_w[j * 65 + 64];
        sFC[r * 16 + j] = fmaxf(acc, 0.0f);
    }
    __syncthreads();
    if (tid < MTILE) {
        float acc = fc2_b[0];
        #pragma unroll
        for (int k = 0; k < 16; ++k)
            acc += sFC[tid * 16 + k] * fc2_w[k];
        out[(size_t)BN * HID + seq0 + tid] = acc;
    }
}

extern "C" void kernel_launch(void* const* d_in, const int* in_sizes, int n_in,
                              void* d_out, int out_size, void* d_ws, size_t ws_size,
                              hipStream_t stream) {
    const float* x     = (const float*)d_in[0];
    const float* Wih0  = (const float*)d_in[1];
    const float* Whh0  = (const float*)d_in[2];
    const float* bih0  = (const float*)d_in[3];
    const float* bhh0  = (const float*)d_in[4];
    const float* Wih1  = (const float*)d_in[5];
    const float* Whh1  = (const float*)d_in[6];
    const float* bih1  = (const float*)d_in[7];
    const float* bhh1  = (const float*)d_in[8];
    const float* fc1_w = (const float*)d_in[9];
    const float* fc1_b = (const float*)d_in[10];
    const float* fc2_w = (const float*)d_in[11];
    const float* fc2_b = (const float*)d_in[12];
    float* out = (float*)d_out;

    dim3 grid(NBLK), block(256);
    hipLaunchKernelGGL(lstm_fused_kernel, grid, block, SMEM_BYTES, stream,
                       x, Wih0, Whh0, bih0, bhh0,
                       Wih1, Whh1, bih1, bhh1,
                       fc1_w, fc1_b, fc2_w, fc2_b, out);
}
